// SAGE_REDDIT_51118700757722
// MI455X (gfx1250) — matrix operand order repaired and checked by reading, not compiled, not run on hardware
//
#include <hip/hip_runtime.h>
#include <hip/hip_bf16.h>
#include <stdint.h>

// ---------- sizes (fixed by the reference) ----------
#define NN 100000   // nodes (divisible by 16)
#define EE 400000   // edges
#define FF 602      // input features
#define FPAD 608    // 602 padded to multiple of 32
#define HH 256      // hidden
#define CC 42       // classes
#define CPAD 48     // 42 padded to multiple of 16
#define KT1 (FPAD / 32)  // 19 K-chunks, layer 1
#define NT1 (HH / 16)    // 16 N-tiles, layer 1
#define KT2 (HH / 32)    // 8  K-chunks, layer 2
#define NT2 (CPAD / 16)  // 3  N-tiles, layer 2

typedef __attribute__((ext_vector_type(16))) __bf16 v16bf;
typedef __attribute__((ext_vector_type(8)))  __bf16 v8bf;
typedef __attribute__((ext_vector_type(8)))  float  v8f;

// ---------- bf16 helpers (round-to-nearest-even) ----------
static __device__ __forceinline__ __bf16 f2bf(float f) {
  union { float f; uint32_t u; } a; a.f = f;
  uint32_t r = (a.u + 0x7FFFu + ((a.u >> 16) & 1u)) >> 16;
  union { uint16_t s; __bf16 b; } o; o.s = (uint16_t)r; return o.b;
}
static __device__ __forceinline__ float bf2f(__bf16 b) {
  union { uint16_t s; __bf16 b; } i; i.b = b;
  union { uint32_t u; float f; } o; o.u = ((uint32_t)i.s) << 16; return o.f;
}

// ---------- generic zero ----------
__global__ void k_zero(float* __restrict__ p, long n) {
  long i = (long)blockIdx.x * blockDim.x + threadIdx.x;
  long st = (long)gridDim.x * blockDim.x;
  for (; i < n; i += st) p[i] = 0.0f;
}

// ---------- degree count (fp32 hardware atomics) ----------
__global__ void k_count(const int* __restrict__ ei, float* __restrict__ cnt) {
  int e = blockIdx.x * blockDim.x + threadIdx.x;
  if (e < EE) unsafeAtomicAdd(&cnt[ei[EE + e]], 1.0f);
}

// ---------- edge scatter-add: one block per edge ----------
__global__ void k_scatter_f32(const int* __restrict__ ei, const float* __restrict__ x,
                              float* __restrict__ sum, int F) {
  int e = blockIdx.x;
  int s = ei[e], d = ei[EE + e];
  const float* xs = x + (long)s * F;
  float* sd = sum + (long)d * F;
  for (int f = threadIdx.x; f < F; f += blockDim.x)
    unsafeAtomicAdd(&sd[f], xs[f]);
}
__global__ void k_scatter_bf16(const int* __restrict__ ei, const __bf16* __restrict__ h,
                               float* __restrict__ sum, int F) {
  int e = blockIdx.x;
  int s = ei[e], d = ei[EE + e];
  const __bf16* hs = h + (long)s * F;
  float* sd = sum + (long)d * F;
  for (int f = threadIdx.x; f < F; f += blockDim.x)
    unsafeAtomicAdd(&sd[f], bf2f(hs[f]));
}

// ---------- convert f32 [rows,k] -> bf16 [rows,kp] zero-padded ----------
__global__ void k_cvt_pad(const float* __restrict__ src, __bf16* __restrict__ dst,
                          int rows, int k, int kp) {
  long n = (long)rows * kp;
  long i = (long)blockIdx.x * blockDim.x + threadIdx.x;
  long st = (long)gridDim.x * blockDim.x;
  for (; i < n; i += st) {
    long r = i / kp; int f = (int)(i - r * kp);
    float v = (f < k) ? src[r * k + f] : 0.0f;
    dst[i] = f2bf(v);
  }
}

// ---------- mean = sum/max(cnt,1) -> bf16 padded ----------
__global__ void k_mean_pad(const float* __restrict__ sum, const float* __restrict__ cnt,
                           __bf16* __restrict__ dst, int rows, int k, int kp) {
  long n = (long)rows * kp;
  long i = (long)blockIdx.x * blockDim.x + threadIdx.x;
  long st = (long)gridDim.x * blockDim.x;
  for (; i < n; i += st) {
    long r = i / kp; int f = (int)(i - r * kp);
    float v = 0.0f;
    if (f < k) v = sum[r * k + f] / fmaxf(cnt[r], 1.0f);
    dst[i] = f2bf(v);
  }
}

// ---------- pack weight f32 [K,Ncols] into per-lane WMMA B fragments ----------
// element i = (((kt*NT + nt)*32 + lane)*16 + e); lane half h splits K:
//   n = nt*16 + (lane&15),  k = kt*32 + 8*(lane>>4) + (e&7) + 16*(e>>3)   (ISA 7.12.2 B layout)
__global__ void k_pack_w(const float* __restrict__ W, __bf16* __restrict__ P,
                         int K, int Ncols, int KT, int NT) {
  long total = (long)KT * NT * 32 * 16;
  long i = (long)blockIdx.x * blockDim.x + threadIdx.x;
  long st = (long)gridDim.x * blockDim.x;
  for (; i < total; i += st) {
    int e    = (int)(i & 15);
    int lane = (int)((i >> 4) & 31);
    long t   = i >> 9;
    int nt   = (int)(t % NT);
    int kt   = (int)(t / NT);
    int n    = nt * 16 + (lane & 15);
    int kk   = kt * 32 + ((lane >> 4) << 3) + (e & 7) + ((e >> 3) << 4);
    float v  = (kk < K && n < Ncols) ? W[(long)kk * Ncols + n] : 0.0f;
    P[i] = f2bf(v);
  }
}

// ---------- A fragment: 16-bit A 16x32; per lane the 8 K-values of each half
// are contiguous in a row-major bf16 row: [kb .. kb+7] and [kb+16 .. kb+23] ----
static __device__ __forceinline__ v16bf load_a(const __bf16* __restrict__ rowptr, int kb) {
  v8bf lo = *(const v8bf*)(rowptr + kb);
  v8bf hi = *(const v8bf*)(rowptr + kb + 16);
  v16bf r;
#pragma unroll
  for (int i = 0; i < 8; ++i) { r[i] = lo[i]; r[i + 8] = hi[i]; }
  return r;
}

// ---------- layer-1 fused GEMM: H = relu(mean@Wl1 + x@Wr1 + b) -> bf16 ----------
// 256 threads = 8 waves; wave w owns N-tiles {2w, 2w+1}; block owns 16 rows.
__global__ void __launch_bounds__(256) k_gemm1(
    const __bf16* __restrict__ Abf,   // mean  [NN, FPAD]
    const __bf16* __restrict__ Xbf,   // x     [NN, FPAD]
    const __bf16* __restrict__ WlP, const __bf16* __restrict__ WrP,
    const float* __restrict__ bias, __bf16* __restrict__ Hout) {
  const int lane = threadIdx.x & 31;
  const int wave = threadIdx.x >> 5;
  const int half = lane >> 4;
  const int m0   = blockIdx.x * 16;
  const int mrow = m0 + (lane & 15);
  const int nt0  = wave * 2;
  const __bf16* arow = Abf + (long)mrow * FPAD;
  const __bf16* xrow = Xbf + (long)mrow * FPAD;
  v8f acc0 = {}; v8f acc1 = {};
  for (int kt = 0; kt < KT1; ++kt) {
    int kb = kt * 32 + 8 * half;
    v16bf am = load_a(arow, kb);
    v16bf ax = load_a(xrow, kb);
    const __bf16* pl = WlP + ((long)(kt * NT1 + nt0)) * 512 + lane * 16;
    const __bf16* pr = WrP + ((long)(kt * NT1 + nt0)) * 512 + lane * 16;
    v16bf bl0 = *(const v16bf*)pl;
    v16bf bl1 = *(const v16bf*)(pl + 512);
    v16bf br0 = *(const v16bf*)pr;
    v16bf br1 = *(const v16bf*)(pr + 512);
    acc0 = __builtin_amdgcn_wmma_f32_16x16x32_bf16(false, am, false, bl0, (short)0, acc0, false, false);
    acc0 = __builtin_amdgcn_wmma_f32_16x16x32_bf16(false, ax, false, br0, (short)0, acc0, false, false);
    acc1 = __builtin_amdgcn_wmma_f32_16x16x32_bf16(false, am, false, bl1, (short)0, acc1, false, false);
    acc1 = __builtin_amdgcn_wmma_f32_16x16x32_bf16(false, ax, false, br1, (short)0, acc1, false, false);
  }
  const int nc0 = nt0 * 16 + (lane & 15);
  const int nc1 = nc0 + 16;
  const float b0 = bias[nc0], b1 = bias[nc1];
#pragma unroll
  for (int r = 0; r < 8; ++r) {
    int m = m0 + r + 8 * half;
    float v0 = acc0[r] + b0; v0 = v0 > 0.0f ? v0 : 0.0f;
    float v1 = acc1[r] + b1; v1 = v1 > 0.0f ? v1 : 0.0f;
    Hout[(long)m * HH + nc0] = f2bf(v0);
    Hout[(long)m * HH + nc1] = f2bf(v1);
  }
}

// ---------- layer-2 fused GEMM: logits = mean2@Wl2 + h@Wr2 + b -> f32 [NN,CPAD] ----
__global__ void __launch_bounds__(96) k_gemm2(
    const __bf16* __restrict__ Mbf,   // mean2 [NN, HH]
    const __bf16* __restrict__ Hbf,   // h     [NN, HH]
    const __bf16* __restrict__ WlP, const __bf16* __restrict__ WrP,
    const float* __restrict__ bias, float* __restrict__ logits) {
  const int lane = threadIdx.x & 31;
  const int nt   = threadIdx.x >> 5;       // 0..2
  const int half = lane >> 4;
  const int m0   = blockIdx.x * 16;
  const int mrow = m0 + (lane & 15);
  const __bf16* arow = Mbf + (long)mrow * HH;
  const __bf16* hrow = Hbf + (long)mrow * HH;
  v8f acc = {};
  for (int kt = 0; kt < KT2; ++kt) {
    int kb = kt * 32 + 8 * half;
    v16bf am = load_a(arow, kb);
    v16bf ah = load_a(hrow, kb);
    v16bf bl = *(const v16bf*)(WlP + ((long)(kt * NT2 + nt)) * 512 + lane * 16);
    v16bf br = *(const v16bf*)(WrP + ((long)(kt * NT2 + nt)) * 512 + lane * 16);
    acc = __builtin_amdgcn_wmma_f32_16x16x32_bf16(false, am, false, bl, (short)0, acc, false, false);
    acc = __builtin_amdgcn_wmma_f32_16x16x32_bf16(false, ah, false, br, (short)0, acc, false, false);
  }
  const int nc = nt * 16 + (lane & 15);
  const float b = (nc < CC) ? bias[nc] : 0.0f;
#pragma unroll
  for (int r = 0; r < 8; ++r) {
    int m = m0 + r + 8 * half;
    logits[(long)m * CPAD + nc] = acc[r] + b;
  }
}

// ---------- log_softmax over 42 cols: one wave32 per row ----------
__global__ void __launch_bounds__(256) k_logsoftmax(const float* __restrict__ logits,
                                                    float* __restrict__ out) {
  const int lane = threadIdx.x & 31;
  const int row  = blockIdx.x * 8 + (threadIdx.x >> 5);
  if (row >= NN) return;
  const float* lr = logits + (long)row * CPAD;
  const float NEG = -__builtin_inff();
  float v0 = (lane < CC) ? lr[lane] : NEG;
  float v1 = (lane + 32 < CC) ? lr[lane + 32] : NEG;
  float mx = fmaxf(v0, v1);
#pragma unroll
  for (int o = 16; o > 0; o >>= 1) mx = fmaxf(mx, __shfl_xor(mx, o, 32));
  float s = ((lane < CC) ? __expf(v0 - mx) : 0.0f) +
            ((lane + 32 < CC) ? __expf(v1 - mx) : 0.0f);
#pragma unroll
  for (int o = 16; o > 0; o >>= 1) s += __shfl_xor(s, o, 32);
  float lse = mx + __logf(s);
  if (lane < CC)      out[(long)row * CC + lane]      = v0 - lse;
  if (lane + 32 < CC) out[(long)row * CC + lane + 32] = v1 - lse;
}

// ---------------------------------------------------------------------------
extern "C" void kernel_launch(void* const* d_in, const int* in_sizes, int n_in,
                              void* d_out, int out_size, void* d_ws, size_t ws_size,
                              hipStream_t stream) {
  (void)in_sizes; (void)n_in; (void)out_size; (void)ws_size;
  const float* x   = (const float*)d_in[0];
  const int*   ei  = (const int*)d_in[1];
  const float* Wl1 = (const float*)d_in[2];
  const float* bl1 = (const float*)d_in[3];
  const float* Wr1 = (const float*)d_in[4];
  const float* Wl2 = (const float*)d_in[5];
  const float* bl2 = (const float*)d_in[6];
  const float* Wr2 = (const float*)d_in[7];
  float* out = (float*)d_out;

  char* ws = (char*)d_ws;
  size_t off = 0;
  auto alloc = [&](size_t bytes) -> char* {
    char* p = ws + off; off += (bytes + 255) & ~(size_t)255; return p;
  };
  float*  cnt    = (float*) alloc((size_t)NN * 4);
  __bf16* xbf    = (__bf16*)alloc((size_t)NN * FPAD * 2);
  __bf16* meanbf = (__bf16*)alloc((size_t)NN * FPAD * 2);
  __bf16* hbf    = (__bf16*)alloc((size_t)NN * HH * 2);
  __bf16* Wl1P   = (__bf16*)alloc((size_t)KT1 * NT1 * 512 * 2);
  __bf16* Wr1P   = (__bf16*)alloc((size_t)KT1 * NT1 * 512 * 2);
  __bf16* Wl2P   = (__bf16*)alloc((size_t)KT2 * NT2 * 512 * 2);
  __bf16* Wr2P   = (__bf16*)alloc((size_t)KT2 * NT2 * 512 * 2);
  // big region reused across phases: sum1 first, then sum2/mean2/logits
  char*   big    = alloc((size_t)NN * FF * 4);          // 240.8 MB
  float*  sum1   = (float*)big;
  float*  sum2   = (float*)big;                         // NN*HH*4 = 102.4 MB
  __bf16* mean2  = (__bf16*)(big + (size_t)NN * HH * 4);  // 51.2 MB
  float*  logits = (float*)(big + (size_t)NN * HH * 4 + (size_t)NN * HH * 2);

  const int T = 256;
  // ---- phase 0: zero + packs + conversions ----
  k_zero<<<4096, T, 0, stream>>>(sum1, (long)NN * FF);
  k_zero<<<(NN + T - 1) / T, T, 0, stream>>>(cnt, (long)NN);
  k_pack_w<<<256, T, 0, stream>>>(Wl1, Wl1P, FF, HH, KT1, NT1);
  k_pack_w<<<256, T, 0, stream>>>(Wr1, Wr1P, FF, HH, KT1, NT1);
  k_pack_w<<<64, T, 0, stream>>>(Wl2, Wl2P, HH, CC, KT2, NT2);
  k_pack_w<<<64, T, 0, stream>>>(Wr2, Wr2P, HH, CC, KT2, NT2);
  k_cvt_pad<<<4096, T, 0, stream>>>(x, xbf, NN, FF, FPAD);
  k_count<<<(EE + T - 1) / T, T, 0, stream>>>(ei, cnt);
  // ---- layer 1 aggregation ----
  k_scatter_f32<<<EE, T, 0, stream>>>(ei, x, sum1, FF);
  k_mean_pad<<<4096, T, 0, stream>>>(sum1, cnt, meanbf, NN, FF, FPAD);
  // ---- layer 1 GEMM (WMMA bf16, fused bias+relu) ----
  k_gemm1<<<NN / 16, 256, 0, stream>>>(meanbf, xbf, Wl1P, Wr1P, bl1, hbf);
  // ---- layer 2 aggregation (sum1 region is dead; reuse as sum2) ----
  k_zero<<<2048, T, 0, stream>>>(sum2, (long)NN * HH);
  k_scatter_bf16<<<EE, T, 0, stream>>>(ei, hbf, sum2, HH);
  k_mean_pad<<<2048, T, 0, stream>>>(sum2, cnt, mean2, NN, HH, HH);
  // ---- layer 2 GEMM + log_softmax ----
  k_gemm2<<<NN / 16, 96, 0, stream>>>(mean2, hbf, Wl2P, Wr2P, bl2, logits);
  k_logsoftmax<<<NN / 8, 256, 0, stream>>>(logits, out);
}
